// ScaledDotProductAttention_1228360646920
// MI455X (gfx1250) — matrix operand order repaired and checked by reading, not compiled, not run on hardware
//
#include <hip/hip_runtime.h>
#include <math.h>

// ---------------------------------------------------------------------------
// Scaled dot-product attention, B=2, H=16, S=2048, D=64 (fp32 in/out),
// returns (output[B,H,S,D], attn_probs[B,H,S,S]) concatenated in d_out.
//
// Two kernels on one stream:
//   1) prep: Q/K/V fp32 -> bf16, pre-swizzled into WMMA fragment layout in ws
//   2) attn: QK^T + softmax + PV on v_wmma_f32_16x16x32_bf16, fragments are
//      single 32B/lane vector loads (no per-element packing VALU).
// One attn block = one (b,h, 16-row query tile); 256 threads = 8 waves.
// ---------------------------------------------------------------------------

typedef __attribute__((ext_vector_type(16))) __bf16 v16bf;
typedef __attribute__((ext_vector_type(16))) unsigned short v16us;
typedef __attribute__((ext_vector_type(8))) float v8f;

#define S_LEN 2048
#define D_DIM 64
#define NHEAD 16
#define QTILE 16
#define NTHREADS 256
#define NBH 32                           // B*H
#define HEAD_ELEMS ((size_t)S_LEN * D_DIM)
#define TENSOR_ELEMS ((size_t)NBH * HEAD_ELEMS)   // 4,194,304 per tensor

// Dynamic LDS layout (floats then u16):
//  sS    : 16*2048 f32 = 32768 floats   (scores -> exp values)
//  sRed  : 256 floats
//  sMax  : 16, sSum : 16
//  sPart : 16*64 = 1024 floats          (PV partial combine)
//  sPbf  : 16*2048 bf16 = 32768 u16     (probs in A-fragment layout)
#define SMEM_F32 (32768 + 256 + 16 + 16 + 1024)        // 34080 floats
#define SMEM_BYTES (SMEM_F32 * 4 + 32768 * 2)          // 201,856 B

__device__ __forceinline__ unsigned short f32_to_bf16(float f) {
  unsigned int u = __builtin_bit_cast(unsigned int, f);
  return (unsigned short)((u + 0x7FFFu + ((u >> 16) & 1u)) >> 16);  // RNE
}

__device__ __forceinline__ v16bf load_frag(const unsigned short* p) {
  return __builtin_bit_cast(v16bf, *(const v16us*)p);
}

// ---------------------------------------------------------------------------
// Prep: scatter fp32 -> bf16 into fragment order [frag][lane][e].
//   Qpre: [bh][qt=128][frag=2][lane=32][e=16]   (A of QK^T: M=q, K=d)
//   Kpre: [bh][kt=128][dc=2] [lane=32][e=16]    (B of QK^T: K=d, N=key)
//   Vpre: [bh][kc=64] [dt=4] [lane=32][e=16]    (B of PV  : K=k, N=d)
// A layout: lane<16 -> M=lane, K in {0..7,16..23}; lane>=16 -> K in {8..15,24..31}
// B layout: lane<16 -> Kdim in {0..7,16..23} (N=lane); lane>=16 -> Kdim in {8..15,24..31} (N=lane-16)
// ---------------------------------------------------------------------------
extern "C" __global__ void __launch_bounds__(NTHREADS)
attn_prep_kernel(const float* __restrict__ Q, const float* __restrict__ K,
                 const float* __restrict__ V, unsigned short* __restrict__ qpre,
                 unsigned short* __restrict__ kpre, unsigned short* __restrict__ vpre) {
  const size_t gid = (size_t)blockIdx.x * NTHREADS + threadIdx.x;
  const size_t tensor = gid / TENSOR_ELEMS;   // 0=Q, 1=K, 2=V
  const size_t idx = gid - tensor * TENSOR_ELEMS;

  const int e = (int)(idx & 15);
  const int lane = (int)((idx >> 4) & 31);
  const size_t f512 = idx >> 9;               // which 512-element fragment

  if (tensor == 0) {
    const int frag = (int)(f512 & 1);
    const int qt = (int)((f512 >> 1) & 127);
    const int bh = (int)(f512 >> 8);
    const int m = lane & 15;
    const int kh = (lane < 16) ? 0 : 8;
    const int d = frag * 32 + kh + ((e < 8) ? e : (e + 8));
    qpre[idx] = f32_to_bf16(Q[((size_t)bh * S_LEN + qt * 16 + m) * D_DIM + d]);
  } else if (tensor == 1) {
    const int dc = (int)(f512 & 1);
    const int kt = (int)((f512 >> 1) & 127);
    const int bh = (int)(f512 >> 8);
    const int key = kt * 16 + (lane & 15);
    const int d = dc * 32 + ((lane < 16) ? 0 : 8) + ((e < 8) ? e : (e + 8));
    kpre[idx] = f32_to_bf16(K[((size_t)bh * S_LEN + key) * D_DIM + d]);
  } else {
    const int dt = (int)(f512 & 3);
    const int kc = (int)((f512 >> 2) & 63);
    const int bh = (int)(f512 >> 8);
    const int d = dt * 16 + (lane & 15);
    const int k = kc * 32 + ((lane < 16) ? 0 : 8) + ((e < 8) ? e : (e + 8));
    vpre[idx] = f32_to_bf16(V[((size_t)bh * S_LEN + k) * D_DIM + d]);
  }
}

// ---------------------------------------------------------------------------
extern "C" __global__ void __launch_bounds__(NTHREADS)
attn_sdpa_kernel(const unsigned short* __restrict__ Qpre,
                 const unsigned short* __restrict__ Kpre,
                 const unsigned short* __restrict__ Vpre,
                 const int* __restrict__ mask,
                 float* __restrict__ out, float* __restrict__ probs) {
  extern __shared__ float smem[];
  float* sS = smem;                          // 16 x 2048 (scores/exp)
  float* sRed = smem + 32768;                // 256
  float* sMax = sRed + 256;                  // 16
  float* sSum = sMax + 16;                   // 16
  float* sPart = sSum + 16;                  // 16 x 64
  unsigned short* sPbf = (unsigned short*)(smem + SMEM_F32);  // 32768 u16

  const int qt = blockIdx.x;
  const int bh = blockIdx.z * NHEAD + blockIdx.y;
  const int qBase = qt * QTILE;
  const int* maskb = mask + (size_t)blockIdx.z * S_LEN;

  const int tid = threadIdx.x;
  const int wave = tid >> 5;
  const int lane = tid & 31;

  // Q A-fragments: two contiguous 32B/lane loads
  const unsigned short* qp =
      Qpre + (((size_t)bh * 128 + qt) * 2) * 512 + (size_t)lane * 16;
  const v16bf aq0 = load_frag(qp);
  const v16bf aq1 = load_frag(qp + 512);

  // ---- stage 1: scores = Q K^T / 8 (each wave: 16 key tiles) ------------
  for (int i = 0; i < 16; ++i) {
    const int kt = wave * 16 + i;
    const unsigned short* kp =
        Kpre + (((size_t)bh * 128 + kt) * 2) * 512 + (size_t)lane * 16;
    const v16bf b0 = load_frag(kp);
    const v16bf b1 = load_frag(kp + 512);
    v8f c = {};
    c = __builtin_amdgcn_wmma_f32_16x16x32_bf16(false, aq0, false, b0,
                                                (short)0, c, false, false);
    c = __builtin_amdgcn_wmma_f32_16x16x32_bf16(false, aq1, false, b1,
                                                (short)0, c, false, false);
    const int key = kt * 16 + (lane & 15);
    const float mval = (maskb[key] == 0) ? -__builtin_inff() : 0.0f;
#pragma unroll
    for (int r = 0; r < 8; ++r) {
      const int m = (lane < 16) ? r : (r + 8);
      sS[m * S_LEN + key] = c[r] * 0.125f + mval;
    }
  }
  __syncthreads();

  // ---- stage 2: row softmax (16 threads per row) ------------------------
  {
    const int row = tid >> 4;
    const int sub = tid & 15;
    float* srow = sS + row * S_LEN;

    float lmax = -__builtin_inff();
    for (int c = sub; c < S_LEN; c += 16) lmax = fmaxf(lmax, srow[c]);
    sRed[tid] = lmax;
    __syncthreads();
    if (sub == 0) {
      float m = -__builtin_inff();
#pragma unroll
      for (int j = 0; j < 16; ++j) m = fmaxf(m, sRed[row * 16 + j]);
      sMax[row] = m;
    }
    __syncthreads();

    const float rmax = sMax[row];
    float lsum = 0.0f;
    for (int c = sub; c < S_LEN; c += 16) {
      const float ev = __expf(srow[c] - rmax);
      srow[c] = ev;
      lsum += ev;
    }
    sRed[tid] = lsum;
    __syncthreads();
    if (sub == 0) {
      float s = 0.0f;
#pragma unroll
      for (int j = 0; j < 16; ++j) s += sRed[row * 16 + j];
      sSum[row] = s;
    }
    __syncthreads();

    // normalize: stream fp32 probs to HBM + bf16 copy into A-fragment layout
    const float inv = 1.0f / sSum[row];
    float* prow = probs + ((size_t)bh * S_LEN + qBase + row) * S_LEN;
    for (int c = sub; c < S_LEN; c += 16) {
      const float p = srow[c] * inv;
      prow[c] = p;
      const int kcIdx = c >> 5;              // 32-wide K chunk
      const int kIn = c & 31;
      const int fl = row + ((kIn & 8) ? 16 : 0);          // fragment lane
      const int fe = (kIn & 7) + ((kIn & 16) ? 8 : 0);    // fragment element
      sPbf[(size_t)kcIdx * 512 + fl * 16 + fe] = f32_to_bf16(p);
    }
  }
  __syncthreads();

  // ---- stage 3: O = P V (8 waves = 4 d-tiles x 2 K-halves) --------------
  const int dTile = wave & 3;
  const int kcStart = (wave >> 2) * 32;      // 32 chunks of 32 keys each
  v8f co = {};
  for (int i = 0; i < 32; ++i) {
    const int kcIdx = kcStart + i;
    const v16bf ap =
        __builtin_bit_cast(v16bf, *(const v16us*)(sPbf + (size_t)kcIdx * 512 +
                                                  (size_t)lane * 16));
    const v16bf bv = load_frag(
        Vpre + (((size_t)bh * 64 + kcIdx) * 4 + dTile) * 512 + (size_t)lane * 16);
    co = __builtin_amdgcn_wmma_f32_16x16x32_bf16(false, ap, false, bv,
                                                 (short)0, co, false, false);
  }
  const int dBase = dTile * 16;
  if (wave >= 4) {
#pragma unroll
    for (int r = 0; r < 8; ++r) {
      const int m = (lane < 16) ? r : (r + 8);
      sPart[m * D_DIM + dBase + (lane & 15)] = co[r];
    }
  }
  __syncthreads();
  if (wave < 4) {
#pragma unroll
    for (int r = 0; r < 8; ++r) {
      const int m = (lane < 16) ? r : (r + 8);
      const float v = co[r] + sPart[m * D_DIM + dBase + (lane & 15)];
      out[((size_t)bh * S_LEN + qBase + m) * D_DIM + dBase + (lane & 15)] = v;
    }
  }
}

// ---------------------------------------------------------------------------
extern "C" void kernel_launch(void* const* d_in, const int* in_sizes, int n_in,
                              void* d_out, int out_size, void* d_ws, size_t ws_size,
                              hipStream_t stream) {
  (void)in_sizes; (void)n_in; (void)out_size; (void)ws_size;

  const float* Q = (const float*)d_in[0];
  const float* K = (const float*)d_in[1];
  const float* V = (const float*)d_in[2];
  const int* mask = (const int*)d_in[3];

  float* out = (float*)d_out;                                  // [2,16,2048,64]
  float* probs = out + (size_t)NBH * S_LEN * D_DIM;            // [2,16,2048,2048]

  unsigned short* qpre = (unsigned short*)d_ws;                // 8 MB
  unsigned short* kpre = qpre + TENSOR_ELEMS;                  // 8 MB
  unsigned short* vpre = kpre + TENSOR_ELEMS;                  // 8 MB

  // 1) pre-swizzle Q/K/V to bf16 fragment layout
  {
    const size_t total = 3 * TENSOR_ELEMS;                     // 12,582,912
    attn_prep_kernel<<<dim3((unsigned)(total / NTHREADS)), dim3(NTHREADS), 0,
                       stream>>>(Q, K, V, qpre, kpre, vpre);
  }

  // 2) attention
  static bool attrSet = false;  // idempotent, deterministic
  if (!attrSet) {
    hipFuncSetAttribute(reinterpret_cast<const void*>(attn_sdpa_kernel),
                        hipFuncAttributeMaxDynamicSharedMemorySize,
                        (int)SMEM_BYTES);
    attrSet = true;
  }
  dim3 grid(S_LEN / QTILE, NHEAD, 2);   // (128, 16, 2)
  attn_sdpa_kernel<<<grid, dim3(NTHREADS), SMEM_BYTES, stream>>>(
      qpre, kpre, vpre, mask, out, probs);
}
